// TransformerBlock_67559835566464
// MI455X (gfx1250) — matrix operand order repaired and checked by reading, not compiled, not run on hardware
//
#include <hip/hip_runtime.h>
#include <math.h>

#define D_RES   1024
#define N_HEADS 16
#define D_HEAD  64
#define D_MLP   4096
#define BATCH   2
#define SEQ     2048
#define ROWS    (BATCH*SEQ)          // 4096
#define EPSLN   1e-5f
#define IGNORE_VAL -100000.0f

typedef __attribute__((ext_vector_type(16))) __bf16 v16bf;
typedef __attribute__((ext_vector_type(8)))  float  v8f;
typedef unsigned short ushort_t;
typedef unsigned int   uint_t;

union Frag16 { v16bf v; uint4 u[2]; };

static __device__ __forceinline__ ushort_t f2bf(float f) {
  union { float f; uint_t u; } x; x.f = f;
  uint_t r = x.u + 0x7FFFu + ((x.u >> 16) & 1u);   // round-to-nearest-even
  return (ushort_t)(r >> 16);
}

// generic __shared__ pointer -> wave-relative LDS byte address (low 32 bits)
static __device__ __forceinline__ uint_t lds_addr(const void* p) {
  return (uint_t)(unsigned long long)p;
}

// async global -> LDS copy of 16 bytes (per lane). INST_OFFSET applies to
// BOTH lds and global address (ISA 10.? async addressing), tracked by ASYNCcnt.
static __device__ __forceinline__ void async_b128(uint_t lds, const void* gptr) {
  asm volatile("global_load_async_to_lds_b128 %0, %1, off"
               :: "v"(lds), "v"(gptr) : "memory");
}
static __device__ __forceinline__ void async_b128_off16(uint_t lds, const void* gptr) {
  asm volatile("global_load_async_to_lds_b128 %0, %1, off offset:16"
               :: "v"(lds), "v"(gptr) : "memory");
}
#define WAIT_ASYNC(n) asm volatile("s_wait_asynccnt " #n ::: "memory")

// ---------------------------------------------------------------------------
// LayerNorm (fp32 in) -> bf16 out.  One block per row of 1024.
// ---------------------------------------------------------------------------
__global__ __launch_bounds__(256)
void ln_to_bf16_kernel(const float* __restrict__ in,
                       const float* __restrict__ w,
                       const float* __restrict__ b,
                       ushort_t* __restrict__ out) {
  const int row = blockIdx.x;
  const int tid = threadIdx.x;     // 256 threads, 4 floats each
  const float4 xv = ((const float4*)(in + (long)row * D_RES))[tid];
  float s  = xv.x + xv.y + xv.z + xv.w;
  float sq = xv.x*xv.x + xv.y*xv.y + xv.z*xv.z + xv.w*xv.w;
  #pragma unroll
  for (int m = 1; m < 32; m <<= 1) {
    s  += __shfl_xor(s,  m);
    sq += __shfl_xor(sq, m);
  }
  __shared__ float rs[8], rq[8];
  const int wv = tid >> 5, ln = tid & 31;
  if (ln == 0) { rs[wv] = s; rq[wv] = sq; }
  __syncthreads();
  s = 0.f; sq = 0.f;
  #pragma unroll
  for (int i = 0; i < 8; i++) { s += rs[i]; sq += rq[i]; }
  const float mean = s * (1.0f / D_RES);
  const float var  = sq * (1.0f / D_RES) - mean * mean;
  const float rstd = 1.0f / sqrtf(var + EPSLN);
  const float4 wv4 = ((const float4*)w)[tid];
  const float4 bv4 = ((const float4*)b)[tid];
  ushort_t* o = out + (long)row * D_RES + tid * 4;
  o[0] = f2bf((xv.x - mean) * rstd * wv4.x + bv4.x);
  o[1] = f2bf((xv.y - mean) * rstd * wv4.y + bv4.y);
  o[2] = f2bf((xv.z - mean) * rstd * wv4.z + bv4.z);
  o[3] = f2bf((xv.w - mean) * rstd * wv4.w + bv4.w);
}

// ---------------------------------------------------------------------------
// Transpose fp32 [B][R][C] -> bf16 [B][C][R]   (weight prep, runs once/call)
// ---------------------------------------------------------------------------
__global__ void transpose_to_bf16_kernel(ushort_t* __restrict__ dst,
                                         const float* __restrict__ src,
                                         int nb, int R, int C) {
  const long total = (long)nb * R * C;
  const long rc = (long)R * C;
  for (long i = (long)blockIdx.x * blockDim.x + threadIdx.x; i < total;
       i += (long)gridDim.x * blockDim.x) {
    long bidx = i / rc;
    long rem  = i - bidx * rc;
    long c = rem / R;
    long r = rem - c * R;
    dst[i] = f2bf(src[bidx * rc + r * C + c]);
  }
}

__global__ void concat_bias_kernel(const float* __restrict__ bq,
                                   const float* __restrict__ bk,
                                   const float* __restrict__ bv,
                                   float* __restrict__ dst) {
  int i = blockIdx.x * blockDim.x + threadIdx.x;
  if (i < 3 * D_RES)
    dst[i] = (i < 1024) ? bq[i] : ((i < 2048) ? bk[i - 1024] : bv[i - 2048]);
}

// ---------------------------------------------------------------------------
// bf16 WMMA GEMM:  C[M,N] = act(A[M,K] @ Bt[N,K]^T + bias) (+resid)
// 256 thr = 8 waves; block tile 128x64; wave tile 32x32 (2x2 WMMA).
// Double-buffered LDS tiles filled by global_load_async_to_lds_b128;
// s_wait_asynccnt 3 keeps the next tile's 3 per-wave async ops in flight
// while guaranteeing the current tile is resident.
// ---------------------------------------------------------------------------
template<bool GELU, bool HAS_RES, bool OUT_F, bool OUT_B>
__global__ __launch_bounds__(256)
void gemm_bf16_wmma_kernel(const ushort_t* __restrict__ A, int lda,
                           const ushort_t* __restrict__ Bt, int ldb,
                           const float* __restrict__ bias,
                           const float* __restrict__ resid,
                           float* __restrict__ outF,
                           ushort_t* __restrict__ outB,
                           int M, int N, int K) {
  __shared__ ushort_t As[2][128 * 40];   // row-major [128][32], padded to 40
  __shared__ ushort_t Bs[2][64 * 40];    // [N=64][K=32], padded to 40

  const int tid  = threadIdx.x;
  const int lane = tid & 31;
  const int wave = tid >> 5;
  const int wm   = wave & 3;          // 4 wave-rows
  const int wn   = wave >> 2;         // 2 wave-cols
  const int mbase = blockIdx.y * 128;
  const int nbase = blockIdx.x * 64;
  const int l16 = lane & 15;
  const int hi  = lane >> 4;

  v8f acc[2][2];
  #pragma unroll
  for (int i = 0; i < 2; i++)
    #pragma unroll
    for (int j = 0; j < 2; j++)
      #pragma unroll
      for (int r = 0; r < 8; r++) acc[i][j][r] = 0.0f;

  const int arow = tid >> 1;          // 0..127
  const int acol = (tid & 1) * 16;    // 0 / 16
  const int brow = tid >> 2;          // 0..63
  const int bcol = (tid & 3) * 8;     // 0,8,16,24

  const ushort_t* agp = A  + (long)(mbase + arow) * lda + acol;
  const ushort_t* bgp = Bt + (long)(nbase + brow) * ldb + bcol;
  const uint_t a_lds0 = lds_addr(&As[0][arow * 40 + acol]);
  const uint_t a_lds1 = lds_addr(&As[1][arow * 40 + acol]);
  const uint_t b_lds0 = lds_addr(&Bs[0][brow * 40 + bcol]);
  const uint_t b_lds1 = lds_addr(&Bs[1][brow * 40 + bcol]);

  const int ntile = K >> 5;

  // prologue: fill buffer 0 with tile 0  (A: 32B as two b128, B: 16B)
  async_b128(a_lds0, agp);
  async_b128_off16(a_lds0, agp);
  async_b128(b_lds0, bgp);

  for (int t = 0; t < ntile; ++t) {
    const int cur = t & 1;
    if (t + 1 < ntile) {
      const ushort_t* ag = agp + (t + 1) * 32;
      const ushort_t* bg = bgp + (t + 1) * 32;
      const uint_t al = cur ? a_lds0 : a_lds1;
      const uint_t bl = cur ? b_lds0 : b_lds1;
      async_b128(al, ag);
      async_b128_off16(al, ag);
      async_b128(bl, bg);
      WAIT_ASYNC(3);                 // tile t resident; tile t+1 in flight
    } else {
      WAIT_ASYNC(0);
    }
    __syncthreads();

    const ushort_t* Ac = &As[cur][0];
    const ushort_t* Bc = &Bs[cur][0];
    Frag16 afr[2], bfr[2];
    #pragma unroll
    for (int mt = 0; mt < 2; mt++) {
      const int row = wm * 32 + mt * 16 + l16;
      const ushort_t* p = Ac + row * 40 + hi * 8;
      afr[mt].u[0] = *(const uint4*)(p);        // K = kb..kb+7
      afr[mt].u[1] = *(const uint4*)(p + 16);   // K = 16+kb..
    }
    #pragma unroll
    for (int nt = 0; nt < 2; nt++) {
      const int n = wn * 32 + nt * 16 + l16;
      const ushort_t* p = Bc + n * 40 + hi * 8;
      bfr[nt].u[0] = *(const uint4*)(p);        // K = kb..kb+7
      bfr[nt].u[1] = *(const uint4*)(p + 16);   // K = 16+kb..
    }
    #pragma unroll
    for (int mt = 0; mt < 2; mt++)
      #pragma unroll
      for (int nt = 0; nt < 2; nt++)
        acc[mt][nt] = __builtin_amdgcn_wmma_f32_16x16x32_bf16(
            false, afr[mt].v, false, bfr[nt].v, (short)0, acc[mt][nt],
            false, false);
    __syncthreads();                 // all waves done reading buf[cur]
  }

  // epilogue
  #pragma unroll
  for (int mt = 0; mt < 2; mt++) {
    #pragma unroll
    for (int nt = 0; nt < 2; nt++) {
      const int col = nbase + wn * 32 + nt * 16 + l16;
      const float bv = bias[col];
      #pragma unroll
      for (int r = 0; r < 8; r++) {
        const int row = mbase + wm * 32 + mt * 16 + hi * 8 + r;
        float v = acc[mt][nt][r] + bv;
        if (GELU) v = 0.5f * v * (1.0f + erff(v * 0.70710678118f));
        const long idx = (long)row * N + col;
        if (HAS_RES) v += resid[idx];
        if (OUT_F) outF[idx] = v;
        if (OUT_B) outB[idx] = f2bf(v);
      }
    }
  }
}

// ---------------------------------------------------------------------------
// Causal flash attention over fused qkv (bf16 [ROWS][3072], q|k|v per head).
// grid = (S/128, B*H); 8 waves, each wave owns 16 q-rows; key blocks of 32.
// K tile staged with async global->LDS copies; V transposed manually.
// ---------------------------------------------------------------------------
__global__ __launch_bounds__(256)
void flash_attn_kernel(const ushort_t* __restrict__ qkv,
                       ushort_t* __restrict__ z) {
  const int bh = blockIdx.y;
  const int bidx = bh >> 4;
  const int h    = bh & 15;
  const int qbase = blockIdx.x * 128;
  const int tid = threadIdx.x, lane = tid & 31, wave = tid >> 5;
  const int l16 = lane & 15, hi = lane >> 4;

  __shared__ ushort_t Ks[32 * 72];        // [key][64], padded rows
  __shared__ ushort_t Vt[64 * 40];        // [dim][32 keys], padded rows
  __shared__ ushort_t Pb[8][16 * 40];     // per-wave P tile [16][32], padded

  const long rowbase = (long)bidx * SEQ;
  const ushort_t* qg = qkv + (rowbase + qbase) * 3072 + h * 64;
  const ushort_t* kg = qkv + rowbase * 3072 + 1024 + h * 64;
  const ushort_t* vg = qkv + rowbase * 3072 + 2048 + h * 64;

  // Q fragments (A layout), rows qbase + wave*16 + l16, K = head dim 64
  Frag16 qa[2];
  {
    const ushort_t* qrow = qg + (long)(wave * 16 + l16) * 3072;
    #pragma unroll
    for (int ks = 0; ks < 2; ks++) {
      qa[ks].u[0] = *(const uint4*)(qrow + 32 * ks + hi * 8);
      qa[ks].u[1] = *(const uint4*)(qrow + 32 * ks + 16 + hi * 8);
    }
  }

  v8f accO[4];
  #pragma unroll
  for (int nt = 0; nt < 4; nt++)
    #pragma unroll
    for (int r = 0; r < 8; r++) accO[nt][r] = 0.0f;
  float mrow[8], lrow[8];
  #pragma unroll
  for (int r = 0; r < 8; r++) { mrow[r] = -1e30f; lrow[r] = 0.0f; }

  const int key_c  = tid >> 3;             // 0..31
  const int part_c = (tid & 7) * 8;        // 0..56
  const uint_t k_lds = lds_addr(&Ks[key_c * 72 + part_c]);

  const int nkb = (qbase >> 5) + 4;        // key blocks covering causal range
  for (int j = 0; j < nkb; j++) {
    const int kb0 = j * 32;
    __syncthreads();
    // stage K tile 32x64 via async copy (1 b128 per lane)
    async_b128(k_lds, kg + (long)(kb0 + key_c) * 3072 + part_c);
    {   // stage V tile transposed -> Vt[dim][key]
      const ushort_t* vp = vg + (long)(kb0 + key_c) * 3072 + part_c;
      #pragma unroll
      for (int i = 0; i < 8; i++) Vt[(part_c + i) * 40 + key_c] = vp[i];
    }
    WAIT_ASYNC(0);
    __syncthreads();

    // scores S = Q @ K^T  (two 16x16 tiles, K-dim = 64 => 2 wmma steps each)
    Frag16 bk[2][2];
    #pragma unroll
    for (int nt = 0; nt < 2; nt++) {
      const int key = nt * 16 + l16;
      #pragma unroll
      for (int ks = 0; ks < 2; ks++) {
        const ushort_t* p = &Ks[key * 72 + 32 * ks + hi * 8];
        bk[nt][ks].u[0] = *(const uint4*)(p);
        bk[nt][ks].u[1] = *(const uint4*)(p + 16);
      }
    }
    v8f s0, s1;
    #pragma unroll
    for (int r = 0; r < 8; r++) { s0[r] = 0.0f; s1[r] = 0.0f; }
    #pragma unroll
    for (int ks = 0; ks < 2; ks++) {
      s0 = __builtin_amdgcn_wmma_f32_16x16x32_bf16(false, qa[ks].v, false, bk[0][ks].v, (short)0, s0, false, false);
      s1 = __builtin_amdgcn_wmma_f32_16x16x32_bf16(false, qa[ks].v, false, bk[1][ks].v, (short)0, s1, false, false);
    }

    // scale + causal mask + online softmax
    float pm[2][8], bmax[8];
    const int qrow0 = qbase + wave * 16 + hi * 8;
    #pragma unroll
    for (int r = 0; r < 8; r++) {
      const int qr = qrow0 + r;
      const float v0 = (kb0 + l16      <= qr) ? s0[r] * 0.125f : IGNORE_VAL;
      const float v1 = (kb0 + 16 + l16 <= qr) ? s1[r] * 0.125f : IGNORE_VAL;
      pm[0][r] = v0; pm[1][r] = v1;
      bmax[r] = fmaxf(v0, v1);
    }
    #pragma unroll
    for (int m = 1; m < 16; m <<= 1)
      #pragma unroll
      for (int r = 0; r < 8; r++) bmax[r] = fmaxf(bmax[r], __shfl_xor(bmax[r], m));

    float scl[8], rsum[8];
    #pragma unroll
    for (int r = 0; r < 8; r++) {
      const float mnew = fmaxf(mrow[r], bmax[r]);
      scl[r] = __expf(mrow[r] - mnew);
      const float p0 = __expf(pm[0][r] - mnew);
      const float p1 = __expf(pm[1][r] - mnew);
      pm[0][r] = p0; pm[1][r] = p1;
      mrow[r] = mnew;
      rsum[r] = p0 + p1;
    }
    #pragma unroll
    for (int m = 1; m < 16; m <<= 1)
      #pragma unroll
      for (int r = 0; r < 8; r++) rsum[r] += __shfl_xor(rsum[r], m);
    #pragma unroll
    for (int r = 0; r < 8; r++) lrow[r] = lrow[r] * scl[r] + rsum[r];
    #pragma unroll
    for (int nt = 0; nt < 4; nt++)
      #pragma unroll
      for (int r = 0; r < 8; r++) accO[nt][r] *= scl[r];

    // P (C-layout) -> LDS -> reload in A-layout for P@V
    ushort_t* pw = &Pb[wave][0];
    #pragma unroll
    for (int r = 0; r < 8; r++) {
      pw[(hi * 8 + r) * 40 + l16]      = f2bf(pm[0][r]);
      pw[(hi * 8 + r) * 40 + 16 + l16] = f2bf(pm[1][r]);
    }
    asm volatile("s_wait_dscnt 0x0" ::: "memory");
    __builtin_amdgcn_wave_barrier();

    Frag16 pa;
    {
      const ushort_t* pr = pw + l16 * 40 + hi * 8;
      pa.u[0] = *(const uint4*)(pr);
      pa.u[1] = *(const uint4*)(pr + 16);
    }
    #pragma unroll
    for (int nt = 0; nt < 4; nt++) {
      Frag16 bv;
      const ushort_t* vp = &Vt[(nt * 16 + l16) * 40 + hi * 8];
      bv.u[0] = *(const uint4*)(vp);
      bv.u[1] = *(const uint4*)(vp + 16);
      accO[nt] = __builtin_amdgcn_wmma_f32_16x16x32_bf16(
          false, pa.v, false, bv.v, (short)0, accO[nt], false, false);
    }
  }

  // normalize + write z (bf16, [row][h*64+dim])
  ushort_t* zr = z + (rowbase + qbase + wave * 16) * 1024 + h * 64;
  #pragma unroll
  for (int r = 0; r < 8; r++) {
    const float inv = 1.0f / lrow[r];
    const int rr = hi * 8 + r;
    #pragma unroll
    for (int nt = 0; nt < 4; nt++)
      zr[(long)rr * 1024 + nt * 16 + l16] = f2bf(accO[nt][r] * inv);
  }
}

// ---------------------------------------------------------------------------
extern "C" void kernel_launch(void* const* d_in, const int* in_sizes, int n_in,
                              void* d_out, int out_size, void* d_ws, size_t ws_size,
                              hipStream_t stream) {
  (void)in_sizes; (void)n_in; (void)out_size; (void)ws_size;
  const float* resid_pre = (const float*)d_in[0];
  const float* ln1_w = (const float*)d_in[1];
  const float* ln1_b = (const float*)d_in[2];
  const float* W_Q  = (const float*)d_in[3];
  const float* b_Q  = (const float*)d_in[4];
  const float* W_K  = (const float*)d_in[5];
  const float* b_K  = (const float*)d_in[6];
  const float* W_V  = (const float*)d_in[7];
  const float* b_V  = (const float*)d_in[8];
  const float* W_O  = (const float*)d_in[9];
  const float* b_O  = (const float*)d_in[10];
  const float* ln2_w = (const float*)d_in[11];
  const float* ln2_b = (const float*)d_in[12];
  const float* W_in  = (const float*)d_in[13];
  const float* b_in  = (const float*)d_in[14];
  const float* W_out = (const float*)d_in[15];
  const float* b_out = (const float*)d_in[16];
  float* out = (float*)d_out;

  char* ws = (char*)d_ws;
  size_t off = 0;
  auto alloc = [&](size_t bytes) -> void* {
    void* p = ws + off;
    off = (off + bytes + 255) & ~(size_t)255;
    return p;
  };
  ushort_t* xb     = (ushort_t*)alloc((size_t)ROWS * 1024 * 2);
  ushort_t* qkvb   = (ushort_t*)alloc((size_t)ROWS * 3072 * 2);
  ushort_t* zb     = (ushort_t*)alloc((size_t)ROWS * 1024 * 2);
  ushort_t* yb     = (ushort_t*)alloc((size_t)ROWS * 1024 * 2);
  ushort_t* hb     = (ushort_t*)alloc((size_t)ROWS * 4096 * 2);
  float*    rm     = (float*)   alloc((size_t)ROWS * 1024 * 4);
  ushort_t* Wqkv_t = (ushort_t*)alloc((size_t)3072 * 1024 * 2);
  ushort_t* Wo_t   = (ushort_t*)alloc((size_t)1024 * 1024 * 2);
  ushort_t* Win_t  = (ushort_t*)alloc((size_t)4096 * 1024 * 2);
  ushort_t* Wout_t = (ushort_t*)alloc((size_t)1024 * 4096 * 2);
  float*    qbias  = (float*)   alloc((size_t)3072 * 4);

  // weight prep (bf16 + [N][K] transpose)
  transpose_to_bf16_kernel<<<1024, 256, 0, stream>>>(Wqkv_t,               W_Q,  16, 1024, 64);
  transpose_to_bf16_kernel<<<1024, 256, 0, stream>>>(Wqkv_t + 1024 * 1024, W_K,  16, 1024, 64);
  transpose_to_bf16_kernel<<<1024, 256, 0, stream>>>(Wqkv_t + 2048 * 1024, W_V,  16, 1024, 64);
  transpose_to_bf16_kernel<<<1024, 256, 0, stream>>>(Wo_t,   W_O,   1, 1024, 1024);
  transpose_to_bf16_kernel<<<2048, 256, 0, stream>>>(Win_t,  W_in,  1, 1024, 4096);
  transpose_to_bf16_kernel<<<2048, 256, 0, stream>>>(Wout_t, W_out, 1, 4096, 1024);
  concat_bias_kernel<<<12, 256, 0, stream>>>(b_Q, b_K, b_V, qbias);

  // LN1
  ln_to_bf16_kernel<<<ROWS, 256, 0, stream>>>(resid_pre, ln1_w, ln1_b, xb);

  // QKV projection: [4096,1024] x [1024,3072] -> bf16 qkv
  gemm_bf16_wmma_kernel<false, false, false, true>
      <<<dim3(3072 / 64, ROWS / 128), 256, 0, stream>>>(
          xb, 1024, Wqkv_t, 1024, qbias, nullptr, nullptr, qkvb,
          ROWS, 3072, 1024);

  // causal flash attention
  flash_attn_kernel<<<dim3(SEQ / 128, BATCH * N_HEADS), 256, 0, stream>>>(qkvb, zb);

  // O projection + residual -> resid_mid (fp32)
  gemm_bf16_wmma_kernel<false, true, true, false>
      <<<dim3(1024 / 64, ROWS / 128), 256, 0, stream>>>(
          zb, 1024, Wo_t, 1024, b_O, resid_pre, rm, nullptr,
          ROWS, 1024, 1024);

  // LN2
  ln_to_bf16_kernel<<<ROWS, 256, 0, stream>>>(rm, ln2_w, ln2_b, yb);

  // MLP in + exact GELU -> bf16 hidden
  gemm_bf16_wmma_kernel<true, false, false, true>
      <<<dim3(4096 / 64, ROWS / 128), 256, 0, stream>>>(
          yb, 1024, Win_t, 1024, b_in, nullptr, nullptr, hb,
          ROWS, 4096, 1024);

  // MLP out + residual -> d_out (fp32)
  gemm_bf16_wmma_kernel<false, true, true, false>
      <<<dim3(1024 / 64, ROWS / 128), 256, 0, stream>>>(
          hb, 4096, Wout_t, 4096, b_out, rm, out, nullptr,
          ROWS, 1024, 4096);
}
